// LogicLayer_65429531787948
// MI455X (gfx1250) — matrix operand order repaired and checked by reading, not compiled, not run on hardware
//
#include <hip/hip_runtime.h>
#include <math.h>

// ---------------------------------------------------------------------------
// Problem: out = sum_g pt[g,s] * gate_g(A,B),  A = softmax(wa)@x, B = softmax(wb)@x
// All 16 gates are bilinear: gate = c0 + cA*A + cB*B + cAB*A*B, so
//   out[s,n] = w0[s] + wA[s]*A + wB[s]*B + wAB[s]*A*B
// => two fused bf16 WMMA GEMMs (M=256,K=256,N=32768) + tiny epilogue.
// x staged to LDS with the Tensor Data Mover (padded rows), double buffered.
// Roofline: 64 MB HBM traffic / 23.3 TB/s ~ 2.75 us; bf16 WMMA keeps the
// 8.6 GFLOP of GEMM work at/below that bound (f32 WMMA K=4 would be ~8x slower).
// ---------------------------------------------------------------------------

#define BATCH 32768
#define PREV  256
#define SIZEM 256

#define BM 128           // block tile M
#define BN 128           // block tile N
#define KC 32            // K chunk (one WMMA K)
#define NKC (PREV / KC)  // 8
#define LDSN 133         // padded LDS row stride in floats (128 + 5, via TDM pad)

typedef __attribute__((ext_vector_type(16))) __bf16 bf16x16;
typedef __attribute__((ext_vector_type(8)))  float  floatx8;
typedef __attribute__((ext_vector_type(4)))  float  floatx4;
typedef __attribute__((ext_vector_type(4)))  unsigned int uintx4;
typedef __attribute__((ext_vector_type(8)))  int    intx8;
typedef __attribute__((ext_vector_type(4)))  int    intx4;

// ---------------------------------------------------------------------------
// Prep 1: row softmax of wa / wb -> bf16 matrices in workspace.
// grid = 512 blocks (0..255: wa rows, 256..511: wb rows), 256 threads.
// ---------------------------------------------------------------------------
__global__ __launch_bounds__(256) void softmax_rows_kernel(
    const float* __restrict__ wa, const float* __restrict__ wb,
    unsigned short* __restrict__ pa, unsigned short* __restrict__ pb)
{
    __shared__ float red[256];
    const int row  = blockIdx.x & 255;
    const bool isB = blockIdx.x >= 256;
    const float* w        = isB ? wb : wa;
    unsigned short* p     = isB ? pb : pa;
    const int t = threadIdx.x;

    float v = w[row * PREV + t];
    red[t] = v; __syncthreads();
    #pragma unroll
    for (int s = 128; s > 0; s >>= 1) {
        if (t < s) red[t] = fmaxf(red[t], red[t + s]);
        __syncthreads();
    }
    const float mx = red[0]; __syncthreads();
    const float e = expf(v - mx);
    red[t] = e; __syncthreads();
    #pragma unroll
    for (int s = 128; s > 0; s >>= 1) {
        if (t < s) red[t] += red[t + s];
        __syncthreads();
    }
    const float pr = e / red[0];
    const __bf16 b = (__bf16)pr;
    p[row * PREV + t] = __builtin_bit_cast(unsigned short, b);
}

// ---------------------------------------------------------------------------
// Prep 2: softmax(wt, axis=0) contracted with constant gate tables.
// coef layout: [w0 | wA | wB | wAB], each 256 floats. 1 block x 256 threads.
// ---------------------------------------------------------------------------
__global__ __launch_bounds__(256) void gate_coeff_kernel(
    const float* __restrict__ wt, float* __restrict__ coef)
{
    const int s = threadIdx.x;
    // gate g = K0 + KA*A + KB*B + KAB*A*B
    const float K0 [16] = {0,0,0,0,0,0,0,0, 1,1,1,1,1,1,1,1};
    const float KA [16] = {0,0,1,1,0,0,1,1, -1,-1,0,0,-1,-1,0,0};
    const float KB [16] = {0,0,0,0,1,1,1,1, -1,-1,-1,-1,0,0,0,0};
    const float KAB[16] = {0,1,-1,0,-1,0,-2,-1, 1,2,0,1,0,1,-1,0};

    float v[16];
    float mx = -3.4e38f;
    #pragma unroll
    for (int g = 0; g < 16; ++g) { v[g] = wt[g * SIZEM + s]; mx = fmaxf(mx, v[g]); }
    float sum = 0.f;
    #pragma unroll
    for (int g = 0; g < 16; ++g) { v[g] = expf(v[g] - mx); sum += v[g]; }
    const float inv = 1.f / sum;
    float c0 = 0.f, cA = 0.f, cB = 0.f, cAB = 0.f;
    #pragma unroll
    for (int g = 0; g < 16; ++g) {
        const float pg = v[g] * inv;
        c0 += pg * K0[g]; cA += pg * KA[g]; cB += pg * KB[g]; cAB += pg * KAB[g];
    }
    coef[          s] = c0;
    coef[SIZEM   + s] = cA;
    coef[2*SIZEM + s] = cB;
    coef[3*SIZEM + s] = cAB;
}

// ---------------------------------------------------------------------------
// TDM: load x tile [KC rows x BN cols] (f32) into LDS with 5-dword padding per
// 128-dword row => LDS row stride 133 floats (conflict-free strided reads).
// ---------------------------------------------------------------------------
__device__ __forceinline__ void tdm_load_tile(unsigned lds_off,
                                              const float* gbase,
                                              int k0, int n0)
{
    const unsigned long long ga =
        (unsigned long long)(const void*)gbase +
        (unsigned long long)((unsigned)(k0 * BATCH + n0)) * 4ull;

    uintx4 g0;
    g0[0] = 1u;                                           // count=1, user D#
    g0[1] = lds_off;                                      // LDS byte address
    g0[2] = (unsigned)ga;                                 // global_addr[31:0]
    g0[3] = ((unsigned)(ga >> 32) & 0x01FFFFFFu)          // global_addr[56:32]
            | 0x80000000u;                                // type=2 ("image")

    intx8 g1;
    g1[0] = (int)((2u << 16)      // data_size = 4 bytes
                | (1u << 20)      // pad_enable
                | (6u << 22)      // pad_interval: 2^(6+1) = 128 dwords
                | (4u << 25));    // pad_amount: 4+1 = 5 dwords
    g1[1] = (int)(((unsigned)(BATCH & 0xFFFF)) << 16);    // tensor_dim0[15:0]
    g1[2] = (int)(((unsigned)BATCH >> 16)                 // tensor_dim0[31:16]
                | (((unsigned)(PREV & 0xFFFF)) << 16));   // tensor_dim1[15:0]
    g1[3] = (int)(((unsigned)PREV >> 16)                  // tensor_dim1[31:16]
                | ((unsigned)BN << 16));                  // tile_dim0 = 128
    g1[4] = (int)((unsigned)KC);                          // tile_dim1 = 32, tile_dim2 = 0
    g1[5] = (int)((unsigned)BATCH);                       // tensor_dim0_stride lo32
    g1[6] = 0;                                            // stride hi bits
    g1[7] = 0;

    const intx4 z4 = {0, 0, 0, 0};
#if __clang_major__ >= 23
    const intx8 z8 = {0, 0, 0, 0, 0, 0, 0, 0};
    __builtin_amdgcn_tensor_load_to_lds(g0, g1, z4, z4, z8, 0);
#else
    __builtin_amdgcn_tensor_load_to_lds(g0, g1, z4, z4, 0);
#endif
}

// ---------------------------------------------------------------------------
// Main fused kernel: block = 128M x 128N, 8 waves (wave32), wave = 32M x 64N.
// Two GEMMs (pa@x, pb@x) share B fragments; bilinear epilogue; f32 out.
// ---------------------------------------------------------------------------
__global__ __launch_bounds__(256) void logic_main_kernel(
    const float* __restrict__ x,
    const unsigned short* __restrict__ paw,
    const unsigned short* __restrict__ pbw,
    const float* __restrict__ coef,
    float* __restrict__ out)
{
    __shared__ float ldsx[2][KC * LDSN];

    const int tid    = threadIdx.x;
    const int lane   = tid & 31;
    const int wave   = tid >> 5;
    const int lane_r = lane & 15;
    const int half   = lane >> 4;
    const int wm     = wave & 3;   // 4 waves along M
    const int wn     = wave >> 2;  // 2 waves along N
    const int blockN = blockIdx.x * BN;
    const int blockM = blockIdx.y * BM;

    floatx8 accA[2][4], accB[2][4];
    #pragma unroll
    for (int mi = 0; mi < 2; ++mi)
        #pragma unroll
        for (int ni = 0; ni < 4; ++ni) {
            accA[mi][ni] = (floatx8)0.0f;
            accB[mi][ni] = (floatx8)0.0f;
        }

    // Prologue: TDM prefetch of chunk 0 (wave 0 drives the DMA engine).
    if (wave == 0)
        tdm_load_tile((unsigned)(size_t)(void*)&ldsx[0][0], x, 0, blockN);

    for (int kc = 0; kc < NKC; ++kc) {
        const int k0 = kc * KC;
        if (wave == 0) {
            if (kc + 1 < NKC) {
                tdm_load_tile((unsigned)(size_t)(void*)&ldsx[(kc + 1) & 1][0],
                              x, k0 + KC, blockN);
                // in-order TENSORcnt: <=1 outstanding means chunk kc has landed
                __builtin_amdgcn_s_wait_tensorcnt((short)1);
            } else {
                __builtin_amdgcn_s_wait_tensorcnt((short)0);
            }
        }
        __syncthreads();

        const float* xb = &ldsx[kc & 1][0];

        // A fragments (bf16, ISA 16-bit A layout: K = 8*half + e / 16+8*half+e-8)
        bf16x16 a_pa[2], a_pb[2];
        #pragma unroll
        for (int mi = 0; mi < 2; ++mi) {
            const int mrow = blockM + wm * 32 + mi * 16 + lane_r;
            const unsigned short* p0 = paw + (size_t)mrow * PREV + k0 + half * 8;
            const unsigned short* p1 = pbw + (size_t)mrow * PREV + k0 + half * 8;
            union { uintx4 u[2]; bf16x16 v; } ua, ub;
            ua.u[0] = *(const uintx4*)(p0);
            ua.u[1] = *(const uintx4*)(p0 + 16);
            ub.u[0] = *(const uintx4*)(p1);
            ub.u[1] = *(const uintx4*)(p1 + 16);
            a_pa[mi] = ua.v;
            a_pb[mi] = ub.v;
        }

        #pragma unroll
        for (int ni = 0; ni < 4; ++ni) {
            const int ncol = wn * 64 + ni * 16 + lane_r;
            // B fragment: lane = column, K = 8*half + e / 16+8*half+e-8; f32->bf16 in VALU
            bf16x16 bf;
            #pragma unroll
            for (int e = 0; e < 16; ++e) {
                const int k = half * 8 + e + (e >= 8 ? 8 : 0);
                bf[e] = (__bf16)xb[k * LDSN + ncol];
            }
            #pragma unroll
            for (int mi = 0; mi < 2; ++mi) {
                accA[mi][ni] = __builtin_amdgcn_wmma_f32_16x16x32_bf16(
                    false, a_pa[mi], false, bf, (short)0, accA[mi][ni], false, false);
                accB[mi][ni] = __builtin_amdgcn_wmma_f32_16x16x32_bf16(
                    false, a_pb[mi], false, bf, (short)0, accB[mi][ni], false, false);
            }
        }
        __syncthreads();   // compute done before this buffer is refilled
    }

    // Epilogue: out = w0 + wA*A + wB*B + wAB*A*B  (coeff frags match D layout)
    #pragma unroll
    for (int mi = 0; mi < 2; ++mi) {
        const int mbase = blockM + wm * 32 + mi * 16 + half * 8;
        union { floatx4 f[2]; floatx8 v; } c0u, cau, cbu, cbau;
        c0u.f[0]  = *(const floatx4*)(coef +             mbase);
        c0u.f[1]  = *(const floatx4*)(coef +             mbase + 4);
        cau.f[0]  = *(const floatx4*)(coef +   SIZEM   + mbase);
        cau.f[1]  = *(const floatx4*)(coef +   SIZEM   + mbase + 4);
        cbu.f[0]  = *(const floatx4*)(coef + 2*SIZEM   + mbase);
        cbu.f[1]  = *(const floatx4*)(coef + 2*SIZEM   + mbase + 4);
        cbau.f[0] = *(const floatx4*)(coef + 3*SIZEM   + mbase);
        cbau.f[1] = *(const floatx4*)(coef + 3*SIZEM   + mbase + 4);

        #pragma unroll
        for (int ni = 0; ni < 4; ++ni) {
            const floatx8 A = accA[mi][ni];
            const floatx8 B = accB[mi][ni];
            const floatx8 r = c0u.v + cau.v * A + cbu.v * B + cbau.v * (A * B);
            const int ncol = blockN + wn * 64 + ni * 16 + lane_r;
            #pragma unroll
            for (int j = 0; j < 8; ++j)
                out[(size_t)(mbase + j) * BATCH + ncol] = r[j];
        }
    }
}

// ---------------------------------------------------------------------------
extern "C" void kernel_launch(void* const* d_in, const int* in_sizes, int n_in,
                              void* d_out, int out_size, void* d_ws, size_t ws_size,
                              hipStream_t stream)
{
    const float* x  = (const float*)d_in[0];  // [PREV, BATCH]
    const float* wa = (const float*)d_in[1];  // [SIZE, PREV]
    const float* wb = (const float*)d_in[2];  // [SIZE, PREV]
    const float* wt = (const float*)d_in[3];  // [16, SIZE]
    float* out = (float*)d_out;               // [SIZE, BATCH]

    unsigned short* pa = (unsigned short*)d_ws;              // bf16 [256,256]
    unsigned short* pb = pa + SIZEM * PREV;                  // bf16 [256,256]
    float* coef = (float*)(pb + SIZEM * PREV);               // 4 x 256 f32

    softmax_rows_kernel<<<dim3(512), dim3(256), 0, stream>>>(wa, wb, pa, pb);
    gate_coeff_kernel<<<dim3(1), dim3(256), 0, stream>>>(wt, coef);

    dim3 grid(BATCH / BN, SIZEM / BM);   // (256, 2)
    logic_main_kernel<<<grid, dim3(256), 0, stream>>>(x, pa, pb, coef, out);
}
